// TopDownParser_4398046511796
// MI455X (gfx1250) — compile-verified, matrix operand order checked by reading, not run on hardware
//
#include <hip/hip_runtime.h>
#include <math.h>

#define T_POS  386
#define HDIM   512
#define G4     2048
#define FF     1024
#define NLAB   128
#define OUTC   129
#define NSPANS 73920   // 384*385/2, divisible by 16

typedef __attribute__((ext_vector_type(16))) __bf16 v16bf;
typedef __attribute__((ext_vector_type(8)))  float  v8f;

union BFrag { v16bf v; unsigned int u[8]; };

__device__ __forceinline__ unsigned short f2bf(float f) {
  unsigned int x = __float_as_uint(f);
  unsigned int r = x + 0x7FFFu + ((x >> 16) & 1u);   // round-to-nearest-even
  return (unsigned short)(r >> 16);
}
__device__ __forceinline__ float sigm(float x) { return 1.0f / (1.0f + __expf(-x)); }

// ---------------------------------------------------------------------------
// K0: span index tables (triu_indices(385, k=1) order) + reset barrier counters
// ---------------------------------------------------------------------------
__global__ void k_init(int* __restrict__ spanL, int* __restrict__ spanR,
                       int* __restrict__ counters) {
  int s = blockIdx.x * blockDim.x + threadIdx.x;
  if (s < 2) counters[s] = 0;
  if (s >= NSPANS) return;
  auto start = [](int l) { return l * 384 - (l * (l - 1)) / 2; };
  int L = (int)floor((769.0 - sqrt(591361.0 - 8.0 * (double)s)) * 0.5);
  if (L < 0) L = 0; if (L > 383) L = 383;
  while (L < 383 && start(L + 1) <= s) ++L;
  while (L > 0 && start(L) > s) --L;
  spanL[s] = L;
  spanR[s] = L + 1 + (s - start(L));
}

// ---------------------------------------------------------------------------
// K1: embedding gather  x[t] = [tag_emb[tag_ids[t]] ; word_emb[word_ids[t]]]
// ---------------------------------------------------------------------------
__global__ void k_embed(const int* __restrict__ tag_ids, const int* __restrict__ word_ids,
                        const float* __restrict__ tag_emb, const float* __restrict__ word_emb,
                        float* __restrict__ x) {
  int idx = blockIdx.x * blockDim.x + threadIdx.x;
  if (idx >= T_POS * HDIM) return;
  int t = idx >> 9, d = idx & 511;
  x[idx] = (d < 128) ? tag_emb[tag_ids[t] * 128 + d]
                     : word_emb[word_ids[t] * 384 + (d - 128)];
}

// ---------------------------------------------------------------------------
// K2: gx = x @ Wih^T + b  for both directions   [386, 2048] each
// ---------------------------------------------------------------------------
__global__ void k_gx(const float* __restrict__ x,
                     const float* __restrict__ Wf, const float* __restrict__ bfv,
                     const float* __restrict__ Wb, const float* __restrict__ bbv,
                     float* __restrict__ gxf, float* __restrict__ gxb) {
  long idx = (long)blockIdx.x * blockDim.x + threadIdx.x;
  const long per = (long)T_POS * G4;
  if (idx >= 2 * per) return;
  int dir = idx >= per;
  long r = idx - (dir ? per : 0);
  int t = (int)(r / G4), row = (int)(r % G4);
  const float4* wr = (const float4*)((dir ? Wb : Wf) + (long)row * HDIM);
  const float4* xr = (const float4*)(x + (long)t * HDIM);
  float acc = 0.f;
  #pragma unroll 4
  for (int k = 0; k < HDIM / 4; ++k) {
    float4 a = xr[k], b = wr[k];
    acc += a.x * b.x + a.y * b.y + a.z * b.z + a.w * b.w;
  }
  acc += (dir ? bbv : bfv)[row];
  (dir ? gxb : gxf)[(long)t * G4 + row] = acc;
}

// ---------------------------------------------------------------------------
// K3: bidirectional LSTM scan. 16 blocks: blk/8 = direction, blk%8 = shard.
// Each shard owns 64 hidden units (their 4 gate rows = 256 rows of Whh).
// Per-step cross-shard sync via atomic counters in workspace.
// ---------------------------------------------------------------------------
__global__ __launch_bounds__(256) void k_scan(
    const float* __restrict__ gxf, const float* __restrict__ gxb,
    const float* __restrict__ Wfh, const float* __restrict__ Wbh,
    float* __restrict__ fwd, float* __restrict__ bwd,
    int* __restrict__ counters) {
  const int tid = threadIdx.x;
  const int blk = blockIdx.x;
  const int dir = blk >> 3;
  const int b   = blk & 7;
  const float* gx  = dir ? gxb : gxf;
  const float* Whh = dir ? Wbh : Wfh;
  float* hout = dir ? bwd : fwd;
  volatile int* cnt = (volatile int*)(counters + dir);

  __shared__ __align__(16) float hbuf[HDIM];
  __shared__ float gbuf[256];

  const int u = tid & 63;           // local hidden unit
  const int q = tid >> 6;           // gate class: 0=i 1=f 2=g 3=o
  const int row = q * 512 + b * 64 + u;     // global gate row
  const float4* wr = (const float4*)(Whh + (long)row * HDIM);

  float c_reg = 0.f;                // valid for tid < 64

  for (int step = 0; step < T_POS; ++step) {
    const int t = dir ? (T_POS - 1 - step) : step;
    if (step == 0) {
      hbuf[tid] = 0.f; hbuf[tid + 256] = 0.f;
    } else {
      const int tprev = dir ? (t + 1) : (t - 1);
      const float* hp = hout + (long)tprev * HDIM;
      hbuf[tid] = hp[tid]; hbuf[tid + 256] = hp[tid + 256];
    }
    __syncthreads();

    float acc = gx[(long)t * G4 + row];
    if (step > 0) {
      const float4* hv4 = (const float4*)hbuf;
      #pragma unroll 4
      for (int k = 0; k < HDIM / 4; ++k) {
        float4 hh = hv4[k], ww = wr[k];
        acc += hh.x * ww.x + hh.y * ww.y + hh.z * ww.z + hh.w * ww.w;
      }
    }
    gbuf[tid] = acc;
    __syncthreads();

    if (tid < 64) {
      float gi = sigm(gbuf[tid]);
      float gf = sigm(gbuf[64 + tid]);
      float gg = tanhf(gbuf[128 + tid]);
      float go = sigm(gbuf[192 + tid]);
      c_reg = gf * c_reg + gi * gg;
      hout[(long)t * HDIM + b * 64 + tid] = go * tanhf(c_reg);
    }
    __threadfence();
    __syncthreads();
    if (tid == 0) {
      atomicAdd((int*)cnt, 1);
      const int target = (step + 1) * 8;
      while (*cnt < target) { }
    }
    __syncthreads();
    __threadfence();
  }
}

// ---------------------------------------------------------------------------
// K4: P/Q tables.  Pf=fwd@W1_top, Pb=bwd@W1_bot, Qf=fwd@V1_top, Qb=bwd@V1_bot
// ---------------------------------------------------------------------------
__global__ void k_tables(const float* __restrict__ fwd, const float* __restrict__ bwd,
                         const float* __restrict__ W1, const float* __restrict__ V1,
                         float* __restrict__ Pf, float* __restrict__ Pb,
                         float* __restrict__ Qf, float* __restrict__ Qb) {
  long idx = (long)blockIdx.x * blockDim.x + threadIdx.x;
  const long per = (long)T_POS * FF;
  if (idx >= 4 * per) return;
  int tbl = (int)(idx / per);
  long r = idx % per;
  int t = (int)(r / FF), j = (int)(r % FF);
  const float* hsrc = (tbl & 1) ? bwd : fwd;
  const float* Wm   = (tbl < 2) ? W1 : V1;
  const int koff = (tbl & 1) ? 512 : 0;
  float acc = 0.f;
  #pragma unroll 4
  for (int k = 0; k < HDIM; ++k)
    acc += hsrc[(long)t * HDIM + k] * Wm[(long)(koff + k) * FF + j];
  float* o = (tbl == 0) ? Pf : (tbl == 1) ? Pb : (tbl == 2) ? Qf : Qb;
  o[(long)t * FF + j] = acc;
}

// ---------------------------------------------------------------------------
// K4b: W2 [1024,128] fp32 -> column-major bf16  W2t[col*1024 + k]
// ---------------------------------------------------------------------------
__global__ void k_w2t(const float* __restrict__ W2, unsigned short* __restrict__ W2t) {
  int idx = blockIdx.x * blockDim.x + threadIdx.x;
  if (idx >= FF * NLAB) return;
  int col = idx >> 10, k = idx & 1023;
  W2t[(long)col * FF + k] = f2bf(W2[(long)k * NLAB + col]);
}

// ---------------------------------------------------------------------------
// K5: fused span kernel. 16 spans/block, 256 thr (8 waves). Phase 1: relu
// hidden (fp32) -> bf16 A-tile in LDS + split-score dot. Phase 2: per-wave
// 16-col WMMA over K=1024 with v_wmma_f32_16x16x32_bf16.
// ---------------------------------------------------------------------------
__global__ __launch_bounds__(256) void k_spans(
    const float* __restrict__ Pf, const float* __restrict__ Pb,
    const float* __restrict__ Qf, const float* __restrict__ Qb,
    const float* __restrict__ b1, const float* __restrict__ c1,
    const unsigned short* __restrict__ W2t, const float* __restrict__ b2,
    const float* __restrict__ V2, const float* __restrict__ c2,
    const int* __restrict__ spanL, const int* __restrict__ spanR,
    float* __restrict__ out) {
  __shared__ __align__(16) unsigned short Abuf[16 * FF];   // 32 KB
  __shared__ float part[16 * 8];
  __shared__ int sL[16], sR[16];

  const int tid  = threadIdx.x;
  const int lane = tid & 31;
  const int w    = tid >> 5;
  const int sbase = blockIdx.x * 16;

  if (tid < 16) { sL[tid] = spanL[sbase + tid]; sR[tid] = spanR[sbase + tid]; }
  __syncthreads();

  float b1r[4], c1r[4], v2r[4];
  #pragma unroll
  for (int kk = 0; kk < 4; ++kk) {
    int jj = kk * 256 + tid;
    b1r[kk] = b1[jj]; c1r[kk] = c1[jj]; v2r[kk] = V2[jj];
  }

  float sp[16];
  for (int s = 0; s < 16; ++s) {
    const int L = sL[s], R = sR[s];
    const float* pfR = Pf + (long)R * FF;       const float* pfL = Pf + (long)L * FF;
    const float* pbL = Pb + (long)(L + 1) * FF; const float* pbR = Pb + (long)(R + 1) * FF;
    const float* qfR = Qf + (long)R * FF;       const float* qfL = Qf + (long)L * FF;
    const float* qbL = Qb + (long)(L + 1) * FF; const float* qbR = Qb + (long)(R + 1) * FF;
    float acc = 0.f;
    #pragma unroll
    for (int kk = 0; kk < 4; ++kk) {
      int jj = kk * 256 + tid;
      float hp = pfR[jj] - pfL[jj] + pbL[jj] - pbR[jj] + b1r[kk];
      Abuf[s * FF + jj] = f2bf(fmaxf(hp, 0.f));
      float hv = qfR[jj] - qfL[jj] + qbL[jj] - qbR[jj] + c1r[kk];
      acc += fmaxf(hv, 0.f) * v2r[kk];
    }
    sp[s] = acc;
  }
  #pragma unroll
  for (int s = 0; s < 16; ++s) {
    float v = sp[s];
    v += __shfl_down(v, 16); v += __shfl_down(v, 8); v += __shfl_down(v, 4);
    v += __shfl_down(v, 2);  v += __shfl_down(v, 1);
    if (lane == 0) part[s * 8 + w] = v;
  }
  __syncthreads();
  if (tid < 16) {
    float v = c2[0];
    #pragma unroll
    for (int i = 0; i < 8; ++i) v += part[tid * 8 + i];
    out[(long)(sbase + tid) * OUTC + NLAB] = v;
  }

  // ----- Phase 2: WMMA. A: lanes 0-15 row m K{0-7,16-23}, lanes 16-31 K{8-15,24-31}
  const int m     = lane & 15;
  const int khalf = lane >> 4;               // 0 or 1
  const int ncol  = w * 16;
  const unsigned int* A32 = (const unsigned int*)Abuf;
  const unsigned int* B32 = (const unsigned int*)(W2t + (long)(ncol + m) * FF);
  const int abase = m * FF;

  v8f cacc = {0.f, 0.f, 0.f, 0.f, 0.f, 0.f, 0.f, 0.f};
  #pragma unroll 4
  for (int kb = 0; kb < FF; kb += 32) {
    BFrag a, bf_;
    const int lo = kb + khalf * 8;
    #pragma unroll
    for (int i = 0; i < 4; ++i) {
      a.u[i]     = A32[(abase + lo + 2 * i) >> 1];
      a.u[4 + i] = A32[(abase + lo + 16 + 2 * i) >> 1];
      bf_.u[i]     = B32[(lo + 2 * i) >> 1];
      bf_.u[4 + i] = B32[(lo + 16 + 2 * i) >> 1];
    }
    cacc = __builtin_amdgcn_wmma_f32_16x16x32_bf16(
        false, a.v, false, bf_.v, (short)0, cacc, false, false);
  }

  const int nOut = ncol + (lane & 15);
  const float bias = b2[nOut];
  #pragma unroll
  for (int r = 0; r < 8; ++r) {
    int M = r + (khalf << 3);                // VGPR r: lanes0-15 M=r, lanes16-31 M=r+8
    out[(long)(sbase + M) * OUTC + nOut] = cacc[r] + bias;
  }
}

// ---------------------------------------------------------------------------
extern "C" void kernel_launch(void* const* d_in, const int* in_sizes, int n_in,
                              void* d_out, int out_size, void* d_ws, size_t ws_size,
                              hipStream_t stream) {
  (void)in_sizes; (void)n_in; (void)out_size; (void)ws_size;
  const int*   tag_ids  = (const int*)d_in[0];
  const int*   word_ids = (const int*)d_in[1];
  const float* tag_emb  = (const float*)d_in[2];
  const float* word_emb = (const float*)d_in[3];
  const float* Wf_ih = (const float*)d_in[4];
  const float* Wf_hh = (const float*)d_in[5];
  const float* bfv   = (const float*)d_in[6];
  const float* Wb_ih = (const float*)d_in[7];
  const float* Wb_hh = (const float*)d_in[8];
  const float* bbv   = (const float*)d_in[9];
  const float* W1 = (const float*)d_in[10];
  const float* b1 = (const float*)d_in[11];
  const float* W2 = (const float*)d_in[12];
  const float* b2 = (const float*)d_in[13];
  const float* V1 = (const float*)d_in[14];
  const float* c1 = (const float*)d_in[15];
  const float* V2 = (const float*)d_in[16];
  const float* c2 = (const float*)d_in[17];
  float* out = (float*)d_out;

  char* ws = (char*)d_ws;
  size_t off = 0;
  auto alloc = [&](size_t bytes) -> char* {
    char* p = ws + off;
    off += (bytes + 255) & ~(size_t)255;
    return p;
  };
  float* x   = (float*)alloc((size_t)T_POS * HDIM * 4);
  float* gxf = (float*)alloc((size_t)T_POS * G4 * 4);
  float* gxb = (float*)alloc((size_t)T_POS * G4 * 4);
  float* fwd = (float*)alloc((size_t)T_POS * HDIM * 4);
  float* bwd = (float*)alloc((size_t)T_POS * HDIM * 4);
  float* Pf  = (float*)alloc((size_t)T_POS * FF * 4);
  float* Pb  = (float*)alloc((size_t)T_POS * FF * 4);
  float* Qf  = (float*)alloc((size_t)T_POS * FF * 4);
  float* Qb  = (float*)alloc((size_t)T_POS * FF * 4);
  unsigned short* W2t = (unsigned short*)alloc((size_t)FF * NLAB * 2);
  int* spanL    = (int*)alloc((size_t)NSPANS * 4);
  int* spanR    = (int*)alloc((size_t)NSPANS * 4);
  int* counters = (int*)alloc(256);

  k_init <<<(NSPANS + 255) / 256, 256, 0, stream>>>(spanL, spanR, counters);
  k_embed<<<(T_POS * HDIM + 255) / 256, 256, 0, stream>>>(tag_ids, word_ids, tag_emb, word_emb, x);
  k_gx   <<<(2 * T_POS * G4 + 255) / 256, 256, 0, stream>>>(x, Wf_ih, bfv, Wb_ih, bbv, gxf, gxb);
  k_scan <<<16, 256, 0, stream>>>(gxf, gxb, Wf_hh, Wb_hh, fwd, bwd, counters);
  k_tables<<<(4 * T_POS * FF + 255) / 256, 256, 0, stream>>>(fwd, bwd, W1, V1, Pf, Pb, Qf, Qb);
  k_w2t  <<<(FF * NLAB + 255) / 256, 256, 0, stream>>>(W2, W2t);
  k_spans<<<NSPANS / 16, 256, 0, stream>>>(Pf, Pb, Qf, Qb, b1, c1, W2t, b2, V2, c2, spanL, spanR, out);
}
